// Attention_35364760715796
// MI455X (gfx1250) — matrix operand order repaired and checked by reading, not compiled, not run on hardware
//
#include <hip/hip_runtime.h>
#include <hip/hip_bf16.h>
#include <math.h>
#include <stdint.h>

// Problem constants (from reference)
#define DM 1024      // d_model = d_q = d_k = d_v
#define SEQ 2048
#define BATCH 4

typedef __attribute__((ext_vector_type(16))) __bf16 v16bf;
typedef __attribute__((ext_vector_type(8)))  float  v8f;
typedef __attribute__((ext_vector_type(4)))  int    v4i_;

#if __has_builtin(__builtin_amdgcn_global_load_async_to_lds_b128) && \
    __has_builtin(__builtin_amdgcn_s_wait_asynccnt)
#define USE_ASYNC_LDS 1
#else
#define USE_ASYNC_LDS 0
#endif

// round-to-nearest-even fp32 -> bf16
__device__ __forceinline__ unsigned short f32_to_bf16(float f) {
    union { float f; unsigned u; } c; c.f = f;
    unsigned u = c.u;
    u += 0x7FFFu + ((u >> 16) & 1u);
    return (unsigned short)(u >> 16);
}
__device__ __forceinline__ unsigned pack_bf16x2(float lo, float hi) {
    return (unsigned)f32_to_bf16(lo) | ((unsigned)f32_to_bf16(hi) << 16);
}

union FragU { v16bf v; uint4 q[2]; };

__device__ __forceinline__ v8f wmma_bf16(const FragU& a, const FragU& b, v8f c) {
    return __builtin_amdgcn_wmma_f32_16x16x32_bf16(false, a.v, false, b.v,
                                                   (short)0, c, false, false);
}

#if USE_ASYNC_LDS
// 16B per lane, global -> LDS, tracked by ASYNCcnt.
// Builtin signature (from hipcc diagnostic): (v4i AS1*, v4i AS3*, imm, imm).
// Generic-LDS addresses carry the LDS byte offset in the low 32 bits
// so the truncating integer cast to AS(3) is valid.
__device__ __forceinline__ void async_b128(const unsigned short* g, unsigned short* l) {
    __builtin_amdgcn_global_load_async_to_lds_b128(
        (__attribute__((address_space(1))) v4i_*)(uintptr_t)g,
        (__attribute__((address_space(3))) v4i_*)(unsigned)(uintptr_t)l,
        0, 0);
}
#endif

// LDS row stride (ushorts): 32 data + 8 pad (breaks bank conflicts, keeps 16B align)
#define LDT 40

// ---------------------------------------------------------------------------
// NT GEMM, all-bf16 inputs: C[m,n] = sum_k A[m,k]*B[n,k]
//   A: MxK bf16 row-major, B: NxK bf16 row-major.
// Block: 256 threads = 8 waves; tile 128x128; K-step 32; double-buffered LDS.
// Wave grid 4x2: wave owns 32 rows x 64 cols -> 2x4 v8f accumulators.
// MODE 0: bf16 store C[m*ldc+n]            (q, k projections)
// MODE 1: bf16 transposed store C[n*ldc+m] (vT)
// MODE 2: fp32 store + causal tile skip    (scores)
// MODE 3: fp32 store + K trimmed to m0+128 (PV; probs upper-tri == 0)
// blockIdx.z = batch, offsets via element strides.
// ---------------------------------------------------------------------------
template <int MODE>
__global__ __launch_bounds__(256)
void gemm_nt_bf16(const unsigned short* __restrict__ A,
                  const unsigned short* __restrict__ Bm,
                  void* __restrict__ Cv, int K,
                  long long strideA, long long strideB, long long strideC,
                  int ldc) {
    const int m0 = blockIdx.y * 128;
    const int n0 = blockIdx.x * 128;

    if (MODE == 2) {
        if (n0 > m0 + 127) return;   // strictly above diagonal: softmax zeroes it
    }

    A  += (long long)blockIdx.z * strideA;
    Bm += (long long)blockIdx.z * strideB;

    __shared__ unsigned short sA[2][128 * LDT];
    __shared__ unsigned short sB[2][128 * LDT];

    const int tid  = threadIdx.x;
    const int lane = tid & 31;
    const int wid  = tid >> 5;            // 0..7
    const int wrow = wid & 3;             // M group of 32
    const int wcol = wid >> 2;            // N group of 64
    const int half = (lane >> 4) & 1;     // lanes 16..31
    const int mr   = lane & 15;

    // staging ownership: thread -> (row, 16-element segment)
    const int srow = tid >> 1;
    const int sseg = (tid & 1) * 16;
    const unsigned short* gA = A  + (long long)(m0 + srow) * K + sseg;
    const unsigned short* gB = Bm + (long long)(n0 + srow) * K + sseg;
    unsigned short* dA0 = &sA[0][srow * LDT + sseg];
    unsigned short* dB0 = &sB[0][srow * LDT + sseg];
    unsigned short* dA1 = &sA[1][srow * LDT + sseg];
    unsigned short* dB1 = &sB[1][srow * LDT + sseg];

    int kend = K;
    if (MODE == 3) { int t = m0 + 128; kend = (t < K) ? t : K; }
    const int nk = kend >> 5;

    v8f acc[2][4] = {};

    // ---- prologue: stage tile 0 into buffer 0 ----
#if USE_ASYNC_LDS
    async_b128(gA,     dA0);  async_b128(gA + 8, dA0 + 8);
    async_b128(gB,     dB0);  async_b128(gB + 8, dB0 + 8);
    __builtin_amdgcn_s_wait_asynccnt(0);
#else
    {
        uint4 ra0 = *(const uint4*)gA, ra1 = *(const uint4*)(gA + 8);
        uint4 rb0 = *(const uint4*)gB, rb1 = *(const uint4*)(gB + 8);
        *(uint4*)dA0 = ra0; *(uint4*)(dA0 + 8) = ra1;
        *(uint4*)dB0 = rb0; *(uint4*)(dB0 + 8) = rb1;
    }
#endif
    __syncthreads();

    for (int kt = 0; kt < nk; ++kt) {
        const int cur = kt & 1;
        const bool more = (kt + 1 < nk);
        const long long koff = (long long)(kt + 1) * 32;

#if USE_ASYNC_LDS
        if (more) {                      // issue async loads for next tile
            unsigned short* nA = cur ? dA0 : dA1;
            unsigned short* nB = cur ? dB0 : dB1;
            async_b128(gA + koff,     nA);  async_b128(gA + koff + 8, nA + 8);
            async_b128(gB + koff,     nB);  async_b128(gB + koff + 8, nB + 8);
        }
#else
        uint4 ra0, ra1, rb0, rb1;
        if (more) {                      // global loads overlap with WMMAs below
            ra0 = *(const uint4*)(gA + koff); ra1 = *(const uint4*)(gA + koff + 8);
            rb0 = *(const uint4*)(gB + koff); rb1 = *(const uint4*)(gB + koff + 8);
        }
#endif

        // ---- compute from buffer `cur` ----
        const unsigned short* pA = sA[cur];
        const unsigned short* pB = sB[cur];

        // A frags (16x32): lanes0-15 VGPR0-3=K0..7, VGPR4-7=K16..23; lanes16-31 +8
        FragU fa0, fa1;
        {
            const unsigned short* a0 = pA + (wrow * 32 + mr) * LDT;
            const unsigned short* a1 = a0 + 16 * LDT;
            fa0.q[0] = *(const uint4*)(a0 + half * 8);
            fa0.q[1] = *(const uint4*)(a0 + 16 + half * 8);
            fa1.q[0] = *(const uint4*)(a1 + half * 8);
            fa1.q[1] = *(const uint4*)(a1 + 16 + half * 8);
        }
        // B frags (32x16): lanes0-15 VGPR0-3=K0..7, VGPR4-7=K16..23; lanes16-31 +8
        #pragma unroll
        for (int nf = 0; nf < 4; ++nf) {
            FragU fb;
            const unsigned short* br = pB + (wcol * 64 + nf * 16 + mr) * LDT + half * 8;
            fb.q[0] = *(const uint4*)(br);
            fb.q[1] = *(const uint4*)(br + 16);
            acc[0][nf] = wmma_bf16(fa0, fb, acc[0][nf]);
            acc[1][nf] = wmma_bf16(fa1, fb, acc[1][nf]);
        }

#if USE_ASYNC_LDS
        __builtin_amdgcn_s_wait_asynccnt(0);
#else
        if (more) {
            unsigned short* nA = cur ? dA0 : dA1;
            unsigned short* nB = cur ? dB0 : dB1;
            *(uint4*)nA = ra0; *(uint4*)(nA + 8) = ra1;
            *(uint4*)nB = rb0; *(uint4*)(nB + 8) = rb1;
        }
#endif
        __syncthreads();
    }

    // ---- epilogue: D layout = VGPR r -> M = r + 8*half, N = lane&15 ----
    #pragma unroll
    for (int mf = 0; mf < 2; ++mf) {
        #pragma unroll
        for (int nf = 0; nf < 4; ++nf) {
            const int n = n0 + wcol * 64 + nf * 16 + mr;
            const int mb = m0 + wrow * 32 + mf * 16 + half * 8;
            #pragma unroll
            for (int r = 0; r < 8; ++r) {
                const int m = mb + r;
                const float val = acc[mf][nf][r];
                if (MODE == 0) {
                    unsigned short* C = (unsigned short*)Cv + (long long)blockIdx.z * strideC;
                    C[(long long)m * ldc + n] = f32_to_bf16(val);
                } else if (MODE == 1) {
                    unsigned short* C = (unsigned short*)Cv + (long long)blockIdx.z * strideC;
                    C[(long long)n * ldc + m] = f32_to_bf16(val);
                } else {
                    float* C = (float*)Cv + (long long)blockIdx.z * strideC;
                    C[(long long)m * ldc + n] = val;
                }
            }
        }
    }
}

// ---------------------------------------------------------------------------
// fp32 -> bf16 bulk convert (8 elements/thread, n multiple of 2048)
// ---------------------------------------------------------------------------
__global__ __launch_bounds__(256)
void cvt_bf16(const float* __restrict__ in, unsigned short* __restrict__ out) {
    const long long i = ((long long)blockIdx.x * 256 + threadIdx.x) * 8;
    float4 a = *(const float4*)(in + i);
    float4 b = *(const float4*)(in + i + 4);
    uint4 p;
    p.x = pack_bf16x2(a.x, a.y); p.y = pack_bf16x2(a.z, a.w);
    p.z = pack_bf16x2(b.x, b.y); p.w = pack_bf16x2(b.z, b.w);
    *(uint4*)(out + i) = p;
}

// ---------------------------------------------------------------------------
// Causal softmax: reads fp32 scores row, writes normalized bf16 probs row,
// zeros the upper triangle (so the PV GEMM is exact). scale = 1/sqrt(d_k).
// ---------------------------------------------------------------------------
__global__ __launch_bounds__(256)
void softmax_causal(float* __restrict__ sc, unsigned short* __restrict__ pb) {
    const int row = blockIdx.x;                 // 0 .. BATCH*SEQ-1
    const int m   = row & (SEQ - 1);
    float* p = sc + (long long)row * SEQ;
    unsigned short* q = pb + (long long)row * SEQ;
    const int nvalid = m + 1;
    const float scale = 0.03125f;               // 1/sqrt(1024)

    __shared__ float red[256];
    const int tid = threadIdx.x;

    float mx = -INFINITY;
    for (int c = tid; c < nvalid; c += 256) mx = fmaxf(mx, p[c] * scale);
    red[tid] = mx; __syncthreads();
    for (int s = 128; s > 0; s >>= 1) {
        if (tid < s) red[tid] = fmaxf(red[tid], red[tid + s]);
        __syncthreads();
    }
    mx = red[0]; __syncthreads();

    float sum = 0.f;
    for (int c = tid; c < nvalid; c += 256) {
        float e = __expf(p[c] * scale - mx);
        p[c] = e;
        sum += e;
    }
    red[tid] = sum; __syncthreads();
    for (int s = 128; s > 0; s >>= 1) {
        if (tid < s) red[tid] += red[tid + s];
        __syncthreads();
    }
    const float inv = 1.0f / red[0];
    for (int c = tid; c < nvalid; c += 256) q[c] = f32_to_bf16(p[c] * inv);
    for (int c = nvalid + tid; c < SEQ; c += 256) q[c] = 0;
}

// ---------------------------------------------------------------------------
extern "C" void kernel_launch(void* const* d_in, const int* in_sizes, int n_in,
                              void* d_out, int out_size, void* d_ws, size_t ws_size,
                              hipStream_t stream) {
    const float* x  = (const float*)d_in[0];   // [B,S,DM]
    const float* Wq = (const float*)d_in[1];   // [DM,DM]
    const float* Wk = (const float*)d_in[2];
    const float* Wv = (const float*)d_in[3];
    float* out = (float*)d_out;                // [B,S,DM]

    // workspace carve
    const long long nX = (long long)BATCH * SEQ * DM;   // 8.39M
    const long long nW = (long long)DM * DM;            // 1.05M
    unsigned short* xb  = (unsigned short*)d_ws;        // [B,S,DM] bf16
    unsigned short* wqb = xb  + nX;                     // [DM,DM]  bf16
    unsigned short* wkb = wqb + nW;
    unsigned short* wvb = wkb + nW;
    unsigned short* qb  = wvb + nW;                     // [B,S,DM] bf16
    unsigned short* kb  = qb  + nX;                     // [B,S,DM] bf16
    unsigned short* vTb = kb  + nX;                     // [B,DM,S] bf16
    unsigned short* pb  = vTb + nX;                     // [B,S,S]  bf16 probs
    float* sc = (float*)(pb + (long long)BATCH * SEQ * SEQ);  // [B,S,S] fp32

    dim3 blk(256);
    const long long sXD = (long long)SEQ * DM;
    const long long sSS = (long long)SEQ * SEQ;

    // bf16 conversions
    cvt_bf16<<<(unsigned)(nX / 2048), blk, 0, stream>>>(x, xb);
    cvt_bf16<<<(unsigned)(nW / 2048), blk, 0, stream>>>(Wq, wqb);
    cvt_bf16<<<(unsigned)(nW / 2048), blk, 0, stream>>>(Wk, wkb);
    cvt_bf16<<<(unsigned)(nW / 2048), blk, 0, stream>>>(Wv, wvb);

    // projections: per batch M=2048, N=1024, K=1024
    dim3 gp(DM / 128, SEQ / 128, BATCH);
    gemm_nt_bf16<0><<<gp, blk, 0, stream>>>(xb, wqb, qb,  DM, sXD, 0, sXD, DM);
    gemm_nt_bf16<0><<<gp, blk, 0, stream>>>(xb, wkb, kb,  DM, sXD, 0, sXD, DM);
    gemm_nt_bf16<1><<<gp, blk, 0, stream>>>(xb, wvb, vTb, DM, sXD, 0, (long long)DM * SEQ, SEQ);

    // scores = q.k^T (fp32 out, causal tiles skipped): M=N=2048, K=1024
    dim3 gs(SEQ / 128, SEQ / 128, BATCH);
    gemm_nt_bf16<2><<<gs, blk, 0, stream>>>(qb, kb, sc, DM, sXD, sXD, sSS, SEQ);

    // causal softmax -> bf16 probs
    softmax_causal<<<BATCH * SEQ, blk, 0, stream>>>(sc, pb);

    // out = P.V == P.(vT)^T : M=2048, N=1024, K=2048 (trimmed to diagonal)
    dim3 go(DM / 128, SEQ / 128, BATCH);
    gemm_nt_bf16<3><<<go, blk, 0, stream>>>(pb, vTb, out, SEQ, sSS, (long long)DM * SEQ, sXD, DM);
}
